// CLAM_SB_38654705664434
// MI455X (gfx1250) — matrix operand order repaired and checked by reading, not compiled, not run on hardware
//
#include <hip/hip_runtime.h>
#include <hip/hip_bf16.h>
#include <math.h>

// ---------------------------------------------------------------------------
// CLAM-SB forward for MI455X (gfx1250, wave32, WMMA bf16 + async-LDS staging)
// ---------------------------------------------------------------------------
typedef __bf16 bf16_t;
typedef __attribute__((ext_vector_type(16))) __bf16 v16bf;
typedef __attribute__((ext_vector_type(8)))  __bf16 v8bf;
typedef __attribute__((ext_vector_type(4)))  __bf16 v4bf;
typedef __attribute__((ext_vector_type(8)))  float  v8f;

#define WMMA_BF16(A, B, C) \
  __builtin_amdgcn_wmma_f32_16x16x32_bf16(false, (A), false, (B), (short)0, (C), false, false)

static __device__ __forceinline__ v16bf cat16(v8bf lo, v8bf hi) {
  return __builtin_shufflevector(lo, hi, 0,1,2,3,4,5,6,7,8,9,10,11,12,13,14,15);
}

// Problem constants
constexpr int Bb   = 4;
constexpr int Nn   = 32768;
constexpr int Ii   = 512;
constexpr int Hh   = 512;
constexpr int Dd   = 256;
constexpr int NCLS = 2;
constexpr int Kk   = 8;
constexpr int ROWS = Bb * Nn;           // 131072

// d_out layout (floats): logits[8] | A[131072] | loss[4] | M[2048]
constexpr int OUT_A    = 8;
constexpr int OUT_LOSS = OUT_A + Bb * Nn;       // 131080
constexpr int OUT_M    = OUT_LOSS + Bb;         // 131084

// workspace layout (bytes)
constexpr size_t WS_H2    = 0;                                   // bf16 [ROWS][512]
constexpr size_t WS_W1T   = WS_H2  + (size_t)ROWS * Hh * 2;      // bf16 [512][512] (transposed)
constexpr size_t WS_WAT   = WS_W1T + (size_t)Hh * Ii * 2;        // bf16 [256][512]
constexpr size_t WS_WBT   = WS_WAT + (size_t)Dd * Hh * 2;        // bf16 [256][512]
constexpr size_t WS_S     = WS_WBT + (size_t)Dd * Hh * 2;        // f32  [ROWS]  raw scores
constexpr size_t WS_MPART = WS_S   + (size_t)ROWS * 4;           // f32  [B][64][512]

// ---------------------------------------------------------------------------
// Kernel 0: convert/transpose weights to bf16  (W1t[n][k], Wat[n][k], Wbt[n][k])
// ---------------------------------------------------------------------------
__global__ void prep_weights(const float* __restrict__ W1, const float* __restrict__ Wa,
                             const float* __restrict__ Wb, bf16_t* __restrict__ W1t,
                             bf16_t* __restrict__ Wat, bf16_t* __restrict__ Wbt) {
  int id = blockIdx.x * blockDim.x + threadIdx.x;        // 524288 total
  if (id < Ii * Hh) {                                    // W1 (512x512) -> W1t
    int n = id / Ii, k = id % Ii;
    W1t[(size_t)n * Ii + k] = (bf16_t)W1[(size_t)k * Hh + n];
  } else if (id < Ii * Hh + Hh * Dd) {                   // Wa (512x256) -> Wat
    int e = id - Ii * Hh;
    int n = e / Hh, k = e % Hh;
    Wat[(size_t)n * Hh + k] = (bf16_t)Wa[(size_t)k * Dd + n];
  } else {                                               // Wb -> Wbt
    int e = id - Ii * Hh - Hh * Dd;
    int n = e / Hh, k = e % Hh;
    Wbt[(size_t)n * Hh + k] = (bf16_t)Wb[(size_t)k * Dd + n];
  }
}

// ---------------------------------------------------------------------------
// Kernel 1: h2 = relu(h @ W1 + b1), stored bf16.
// 512 thr (16 waves), tile 64 rows x 512 cols (full H), grid = ROWS/64.
// Double-buffered LDS A-tile (f32 -> bf16), register-group B prefetch.
// ---------------------------------------------------------------------------
__global__ void __launch_bounds__(512)
gemm1_relu(const float* __restrict__ h, const bf16_t* __restrict__ W1t,
           const float* __restrict__ b1, bf16_t* __restrict__ h2) {
  __shared__ alignas(16) bf16_t Ash[2][64][32];          // 2 x 4 KB

  const int tid  = threadIdx.x;
  const int w    = tid >> 5, lane = tid & 31;
  const int r    = w & 3, cgp = w >> 2;                  // row tile, col group (0..3)
  const int m0   = blockIdx.x * 64;
  const int cb   = cgp * 128;
  const int la15 = lane & 15, half = lane >> 4;

  v8f acc[8];
  const v8f vzero = {0,0,0,0,0,0,0,0};
#pragma unroll
  for (int t = 0; t < 8; ++t) acc[t] = vzero;

  const int srr = tid >> 3, scc = (tid & 7) * 4;         // staging: 4 floats / thread

  { // stage tile 0
    const float4 f = *(const float4*)&h[(size_t)(m0 + srr) * Ii + scc];
    v4bf o = {(bf16_t)f.x, (bf16_t)f.y, (bf16_t)f.z, (bf16_t)f.w};
    *(v4bf*)&Ash[0][srr][scc] = o;
  }
  __syncthreads();

  for (int ks = 0; ks < Ii / 32; ++ks) {
    const int k0 = ks * 32;
    const int cur = ks & 1, nxt = cur ^ 1;
    if (ks + 1 < Ii / 32) {                              // stage next tile
      const float4 f = *(const float4*)&h[(size_t)(m0 + srr) * Ii + k0 + 32 + scc];
      v4bf o = {(bf16_t)f.x, (bf16_t)f.y, (bf16_t)f.z, (bf16_t)f.w};
      *(v4bf*)&Ash[nxt][srr][scc] = o;
    }

    v8bf alo = *(const v8bf*)&Ash[cur][r * 16 + la15][half * 16];
    v8bf ahi = *(const v8bf*)&Ash[cur][r * 16 + la15][half * 16 + 8];
    v16bf afrag = cat16(alo, ahi);

    v16bf bf[8];                                         // prefetch all B frags
#pragma unroll
    for (int t = 0; t < 8; ++t)
      bf[t] = *(const v16bf*)&W1t[(size_t)(cb + t * 16 + la15) * Ii + k0 + half * 16];
#pragma unroll
    for (int t = 0; t < 8; ++t)
      acc[t] = WMMA_BF16(afrag, bf[t], acc[t]);

    __syncthreads();
  }

  // epilogue: C layout VGPR i -> row i + 8*half; col = lane&15
  const int rbase = r * 16 + half * 8;
#pragma unroll
  for (int t = 0; t < 8; ++t) {
    int col = cb + t * 16 + la15;
    float bias = b1[col];
#pragma unroll
    for (int i = 0; i < 8; ++i) {
      float v = acc[t][i] + bias;
      v = v > 0.f ? v : 0.f;
      h2[(size_t)(m0 + rbase + i) * Hh + col] = (bf16_t)v;
    }
  }
}

// ---------------------------------------------------------------------------
// Kernel 2: a = tanh(h2@Wa+ba), g = sigmoid(h2@Wb+bb), s = (a*g)·Wc + bc
// 512 thr (16 waves), tile 64 rows x 256 cols. A-tile staged via CDNA5
// global_load_async_to_lds (ASYNCcnt), double buffered.
// ---------------------------------------------------------------------------
__global__ void __launch_bounds__(512)
gemm2_gate_score(const bf16_t* __restrict__ h2, const bf16_t* __restrict__ Wat,
                 const bf16_t* __restrict__ Wbt, const float* __restrict__ ba,
                 const float* __restrict__ bb, const float* __restrict__ Wc,
                 const float* __restrict__ bc, float* __restrict__ sbuf) {
  __shared__ alignas(16) bf16_t Hsh[2][64][32];          // 2 x 4 KB async-staged
  __shared__ float red[64][64];                          // 16 KB deterministic reduce

  const int tid  = threadIdx.x;
  const int w    = tid >> 5, lane = tid & 31;
  const int r    = w & 3, cgp = w >> 2;                  // 4 row tiles x 4 col groups
  const int m0   = blockIdx.x * 64;
  const int la15 = lane & 15, half = lane >> 4;

  v8f ca[4], cg[4];
  const v8f vzero = {0,0,0,0,0,0,0,0};
#pragma unroll
  for (int t = 0; t < 4; ++t) { ca[t] = vzero; cg[t] = vzero; }

  // async staging geometry: tile row = tid>>3 (64B rows), 8B chunk = tid&7
  const int srr = tid >> 3;
  const int scb = (tid & 7) * 8;                         // byte offset in row
  const unsigned ldsBase = (unsigned)(uintptr_t)(&Hsh[0][0][0]);
  const unsigned ldsOff0 = ldsBase + (unsigned)(tid * 8);
  const unsigned ldsOff1 = ldsBase + (unsigned)(64 * 32 * 2) + (unsigned)(tid * 8);
  const char* gRow = (const char*)(h2 + (size_t)(m0 + srr) * Hh) + scb;

  { // issue async copy of tile 0 (k0 = 0)
    asm volatile("global_load_async_to_lds_b64 %0, %1, off"
                 :: "v"(ldsOff0), "v"(gRow) : "memory");
  }
  asm volatile("s_wait_asynccnt 0x0" ::: "memory");
  __syncthreads();

  for (int ks = 0; ks < Hh / 32; ++ks) {
    const int k0 = ks * 32;
    const int cur = ks & 1;
    if (ks + 1 < Hh / 32) {                              // async-stage next tile
      unsigned ldsN = (cur == 0) ? ldsOff1 : ldsOff0;
      const char* gN = gRow + (size_t)(k0 + 32) * 2;
      asm volatile("global_load_async_to_lds_b64 %0, %1, off"
                   :: "v"(ldsN), "v"(gN) : "memory");
    }

    v8bf alo = *(const v8bf*)&Hsh[cur][r * 16 + la15][half * 16];
    v8bf ahi = *(const v8bf*)&Hsh[cur][r * 16 + la15][half * 16 + 8];
    v16bf afrag = cat16(alo, ahi);

    v16bf bA[4], bB[4];                                  // prefetch weight frags
#pragma unroll
    for (int t = 0; t < 4; ++t) {
      int col = cgp * 64 + t * 16 + la15;
      bA[t] = *(const v16bf*)&Wat[(size_t)col * Hh + k0 + half * 16];
      bB[t] = *(const v16bf*)&Wbt[(size_t)col * Hh + k0 + half * 16];
    }
#pragma unroll
    for (int t = 0; t < 4; ++t) {
      ca[t] = WMMA_BF16(afrag, bA[t], ca[t]);
      cg[t] = WMMA_BF16(afrag, bB[t], cg[t]);
    }

    asm volatile("s_wait_asynccnt 0x0" ::: "memory");
    __syncthreads();
  }

  // per-lane partial row sums of a*g*Wc
  float part[8];
#pragma unroll
  for (int i = 0; i < 8; ++i) part[i] = 0.f;
#pragma unroll
  for (int t = 0; t < 4; ++t) {
    int col = cgp * 64 + t * 16 + la15;
    float bav = ba[col], bbv = bb[col], wcv = Wc[col];
#pragma unroll
    for (int i = 0; i < 8; ++i) {
      float av = tanhf(ca[t][i] + bav);
      float gv = 1.f / (1.f + expf(-(cg[t][i] + bbv)));
      part[i] += av * gv * wcv;
    }
  }
  const int rb = r * 16 + half * 8;
  const int slot = cgp * 16 + la15;                      // 64 slots
#pragma unroll
  for (int i = 0; i < 8; ++i) red[rb + i][slot] = part[i];
  __syncthreads();

  if (tid < 64) {
    float s = bc[0];
#pragma unroll
    for (int j = 0; j < 64; ++j) s += red[tid][j];
    sbuf[m0 + tid] = s;
  }
}

// ---------------------------------------------------------------------------
// Kernel 3: softmax over N per batch -> A (d_out).  Deterministic tree reduce.
// ---------------------------------------------------------------------------
__global__ void __launch_bounds__(1024)
softmax_rows(const float* __restrict__ sbuf, float* __restrict__ out) {
  __shared__ float red[1024];
  const int b = blockIdx.x, tid = threadIdx.x;
  const float* s = sbuf + (size_t)b * Nn;

  float m = -3.402823466e+38f;
  for (int j = 0; j < Nn / 1024; ++j) {
    float v = s[j * 1024 + tid];
    m = v > m ? v : m;
  }
  red[tid] = m; __syncthreads();
  for (int off = 512; off > 0; off >>= 1) {
    if (tid < off) { float o = red[tid + off]; if (o > red[tid]) red[tid] = o; }
    __syncthreads();
  }
  const float gmax = red[0]; __syncthreads();

  float acc = 0.f;
  for (int j = 0; j < Nn / 1024; ++j) acc += expf(s[j * 1024 + tid] - gmax);
  red[tid] = acc; __syncthreads();
  for (int off = 512; off > 0; off >>= 1) {
    if (tid < off) red[tid] += red[tid + off];
    __syncthreads();
  }
  const float inv = 1.f / red[0];

  float* A = out + OUT_A + (size_t)b * Nn;
  for (int j = 0; j < Nn / 1024; ++j) {
    int n = j * 1024 + tid;
    A[n] = expf(s[n] - gmax) * inv;
  }
}

// ---------------------------------------------------------------------------
// Kernel 4: Mpart[b][ch][c] = sum_{n in chunk} A[b][n] * h2[b][n][c]
// ---------------------------------------------------------------------------
__global__ void __launch_bounds__(512)
weighted_sum_part(const float* __restrict__ out, const bf16_t* __restrict__ h2,
                  float* __restrict__ Mpart) {
  const int b = blockIdx.x, ch = blockIdx.y, c = threadIdx.x;
  const float* A = out + OUT_A + (size_t)b * Nn + ch * 512;
  const bf16_t* hp = h2 + ((size_t)b * Nn + (size_t)ch * 512) * Hh;
  float acc = 0.f;
  for (int n = 0; n < 512; ++n)
    acc += A[n] * (float)hp[(size_t)n * Hh + c];
  Mpart[((size_t)b * 64 + ch) * 512 + c] = acc;
}

// ---------------------------------------------------------------------------
// Kernel 5: M = sum partials (deterministic), logits = M @ Wcls + bcls
// ---------------------------------------------------------------------------
__global__ void __launch_bounds__(512)
finalize_M_logits(const float* __restrict__ Mpart, const float* __restrict__ Wcls,
                  const float* __restrict__ bcls, float* __restrict__ out) {
  __shared__ float Msh[512];
  const int b = blockIdx.x, c = threadIdx.x;
  float m = 0.f;
  for (int j = 0; j < 64; ++j) m += Mpart[((size_t)b * 64 + j) * 512 + c];
  out[OUT_M + (size_t)b * Hh + c] = m;
  Msh[c] = m;
  __syncthreads();
  if (c < NCLS) {
    float lg = bcls[c];
    for (int hh = 0; hh < Hh; ++hh) lg += Msh[hh] * Wcls[hh * NCLS + c];
    out[b * NCLS + c] = lg;
  }
}

// ---------------------------------------------------------------------------
// Kernel 6: iterative top-8 / bottom-8 over scores + instance CE losses
// ---------------------------------------------------------------------------
__global__ void __launch_bounds__(256)
topk_inst_loss(const float* __restrict__ sbuf, const bf16_t* __restrict__ h2,
               const float* __restrict__ Winst, const float* __restrict__ binst,
               const int* __restrict__ label, float* __restrict__ out) {
  __shared__ unsigned mask[Nn / 32];                     // 4 KB bitmask
  __shared__ float bv[256];
  __shared__ int   bi[256];
  __shared__ int   pid[Kk], nid[Kk];
  __shared__ float lg[NCLS][2 * Kk][2];

  const int b = blockIdx.x, tid = threadIdx.x;
  const float* s = sbuf + (size_t)b * Nn;

  for (int j = tid; j < Nn / 32; j += 256) mask[j] = 0u;
  __syncthreads();

  // ---- top-K (largest, ties -> lower index) ----
  for (int k = 0; k < Kk; ++k) {
    float best = -3.402823466e+38f; int bidx = 1 << 30;
    for (int j = 0; j < Nn / 256; ++j) {
      int n = tid * (Nn / 256) + j;
      if ((mask[n >> 5] >> (n & 31)) & 1u) continue;
      float v = s[n];
      if (v > best || (v == best && n < bidx)) { best = v; bidx = n; }
    }
    bv[tid] = best; bi[tid] = bidx; __syncthreads();
    for (int off = 128; off > 0; off >>= 1) {
      if (tid < off) {
        float v2 = bv[tid + off]; int i2 = bi[tid + off];
        if (v2 > bv[tid] || (v2 == bv[tid] && i2 < bi[tid])) { bv[tid] = v2; bi[tid] = i2; }
      }
      __syncthreads();
    }
    if (tid == 0) { pid[k] = bi[0]; mask[bi[0] >> 5] |= 1u << (bi[0] & 31); }
    __syncthreads();
  }

  for (int j = tid; j < Nn / 32; j += 256) mask[j] = 0u;
  __syncthreads();

  // ---- bottom-K (smallest, ties -> lower index) ----
  for (int k = 0; k < Kk; ++k) {
    float best = 3.402823466e+38f; int bidx = 1 << 30;
    for (int j = 0; j < Nn / 256; ++j) {
      int n = tid * (Nn / 256) + j;
      if ((mask[n >> 5] >> (n & 31)) & 1u) continue;
      float v = s[n];
      if (v < best || (v == best && n < bidx)) { best = v; bidx = n; }
    }
    bv[tid] = best; bi[tid] = bidx; __syncthreads();
    for (int off = 128; off > 0; off >>= 1) {
      if (tid < off) {
        float v2 = bv[tid + off]; int i2 = bi[tid + off];
        if (v2 < bv[tid] || (v2 == bv[tid] && i2 < bi[tid])) { bv[tid] = v2; bi[tid] = i2; }
      }
      __syncthreads();
    }
    if (tid == 0) { nid[k] = bi[0]; mask[bi[0] >> 5] |= 1u << (bi[0] & 31); }
    __syncthreads();
  }

  // ---- 64 instance logits: 2 classes x 16 instances x 2 cols ----
  if (tid < 64) {
    int cls = tid >> 5, j = (tid >> 1) & 15, col = tid & 1;
    int row = (j < Kk) ? pid[j] : nid[j - Kk];
    const bf16_t* hr = h2 + ((size_t)b * Nn + row) * Hh;
    float acc = binst[cls * 2 + col];
    for (int hh = 0; hh < Hh; ++hh)
      acc += (float)hr[hh] * Winst[((size_t)cls * Hh + hh) * 2 + col];
    lg[cls][j][col] = acc;
  }
  __syncthreads();

  if (tid == 0) {
    const int lab = *label;
    float total = 0.f;
    for (int cls = 0; cls < NCLS; ++cls) {
      if (cls == lab) {                                  // in-class: 8 pos + 8 neg
        float sum = 0.f;
        for (int j = 0; j < 2 * Kk; ++j) {
          float l0 = lg[cls][j][0], l1 = lg[cls][j][1];
          float mx = fmaxf(l0, l1);
          float lse = mx + logf(expf(l0 - mx) + expf(l1 - mx));
          sum += lse - ((j < Kk) ? l1 : l0);
        }
        total += sum / (2.f * Kk);
      } else {                                           // out-class: 8 pos, target 0
        float sum = 0.f;
        for (int j = 0; j < Kk; ++j) {
          float l0 = lg[cls][j][0], l1 = lg[cls][j][1];
          float mx = fmaxf(l0, l1);
          float lse = mx + logf(expf(l0 - mx) + expf(l1 - mx));
          sum += lse - l0;
        }
        total += sum / (float)Kk;
      }
    }
    out[OUT_LOSS + b] = total / (float)NCLS;
  }
}

// ---------------------------------------------------------------------------
extern "C" void kernel_launch(void* const* d_in, const int* in_sizes, int n_in,
                              void* d_out, int out_size, void* d_ws, size_t ws_size,
                              hipStream_t stream) {
  const float* h     = (const float*)d_in[0];
  const float* W1    = (const float*)d_in[1];
  const float* b1    = (const float*)d_in[2];
  const float* Wa    = (const float*)d_in[3];
  const float* ba    = (const float*)d_in[4];
  const float* Wb    = (const float*)d_in[5];
  const float* bb    = (const float*)d_in[6];
  const float* Wc    = (const float*)d_in[7];
  const float* bc    = (const float*)d_in[8];
  const float* Wcls  = (const float*)d_in[9];
  const float* bcls  = (const float*)d_in[10];
  const float* Winst = (const float*)d_in[11];
  const float* binst = (const float*)d_in[12];
  const int*   label = (const int*)d_in[13];

  float* out = (float*)d_out;
  char*  ws  = (char*)d_ws;
  bf16_t* h2    = (bf16_t*)(ws + WS_H2);
  bf16_t* W1t   = (bf16_t*)(ws + WS_W1T);
  bf16_t* Wat   = (bf16_t*)(ws + WS_WAT);
  bf16_t* Wbt   = (bf16_t*)(ws + WS_WBT);
  float*  sbuf  = (float*)(ws + WS_S);
  float*  Mpart = (float*)(ws + WS_MPART);

  prep_weights<<<dim3((Ii * Hh + 2 * Hh * Dd) / 256), dim3(256), 0, stream>>>(
      W1, Wa, Wb, W1t, Wat, Wbt);

  gemm1_relu<<<dim3(ROWS / 64), dim3(512), 0, stream>>>(h, W1t, b1, h2);

  gemm2_gate_score<<<dim3(ROWS / 64), dim3(512), 0, stream>>>(
      h2, Wat, Wbt, ba, bb, Wc, bc, sbuf);

  softmax_rows<<<dim3(Bb), dim3(1024), 0, stream>>>(sbuf, out);

  weighted_sum_part<<<dim3(Bb, 64), dim3(512), 0, stream>>>(out, h2, Mpart);

  finalize_M_logits<<<dim3(Bb), dim3(512), 0, stream>>>(Mpart, Wcls, bcls, out);

  topk_inst_loss<<<dim3(Bb), dim3(256), 0, stream>>>(sbuf, h2, Winst, binst, label, out);
}
